// SparseDiffAttn_40389872452031
// MI455X (gfx1250) — matrix operand order repaired and checked by reading, not compiled, not run on hardware
//
#include <hip/hip_runtime.h>

typedef __attribute__((ext_vector_type(16))) _Float16 v16h;
typedef __attribute__((ext_vector_type(8)))  _Float16 v8h;
typedef __attribute__((ext_vector_type(8)))  float    v8f;
typedef __attribute__((ext_vector_type(4)))  unsigned u32x4;
typedef __attribute__((ext_vector_type(8)))  int      i32x8;
typedef __attribute__((ext_vector_type(4)))  int      i32x4;

#if defined(__has_builtin)
#if __has_builtin(__builtin_amdgcn_global_load_async_to_lds_b128)
#define ASYNC_LDS 1
#endif
#endif

namespace {

constexpr int H   = 12;
constexpr int S   = 3840;
constexpr int D   = 128;
constexpr int BM  = 192;
constexpr int QG  = S / BM;   // 20 query groups
constexpr int BN  = 64;       // keys per chunk
constexpr int NW  = 6;        // waves per block (192 threads, wave32)
constexpr int LDK = D + 8;    // padded LDS stride (halves): 272B rows -> conflict-free B-frag reads
constexpr int LDV = BN + 8;   // padded stride for transposed V
constexpr int LDP = BN + 8;   // padded stride for staged P
// 1/sqrt(128) * log2(e): fold softmax scale + exp->exp2 conversion into Q.
constexpr float SCALE_LOG2E = 0.08838834764831845f * 1.4426950408889634f;

__device__ inline v8f wmma16(v16h a, v16h b, v8f c) {
  // emits v_wmma_f32_16x16x32_f16
  return __builtin_amdgcn_wmma_f32_16x16x32_f16(false, a, false, b, (short)0, c,
                                                false, false);
}

__device__ inline float redmax16(float x) {
  x = fmaxf(x, __shfl_xor(x, 1, 32));
  x = fmaxf(x, __shfl_xor(x, 2, 32));
  x = fmaxf(x, __shfl_xor(x, 4, 32));
  x = fmaxf(x, __shfl_xor(x, 8, 32));
  return x;
}
__device__ inline float redsum16(float x) {
  x += __shfl_xor(x, 1, 32);
  x += __shfl_xor(x, 2, 32);
  x += __shfl_xor(x, 4, 32);
  x += __shfl_xor(x, 8, 32);
  return x;
}

// A fragment (16x32 f16, MxK), ISA layout: lane m=L&15; halves 0..7 -> K=kb..kb+7,
// halves 8..15 -> K=kb+16..kb+23, kb = k0 + 8*(L>=16).
__device__ inline v16h load_a_global(const float* base, int row0, int k0,
                                     int lane, float scl) {
  const float* p = base + (size_t)(row0 + (lane & 15)) * D + k0 + ((lane >> 4) << 3);
  v16h a;
#pragma unroll
  for (int i = 0; i < 8; ++i) a[i] = (_Float16)(p[i] * scl);
#pragma unroll
  for (int i = 0; i < 8; ++i) a[8 + i] = (_Float16)(p[16 + i] * scl);
  return a;
}

__device__ inline v16h load_a_lds(const _Float16* m, int row0, int k0, int ld,
                                  int lane) {
  const _Float16* p = m + (row0 + (lane & 15)) * ld + k0 + ((lane >> 4) << 3);
  v16h a;
#pragma unroll
  for (int i = 0; i < 8; ++i) a[i] = p[i];
#pragma unroll
  for (int i = 0; i < 8; ++i) a[8 + i] = p[16 + i];
  return a;
}

// B fragment (32x16 f16, KxN): element (k,n) = m[(n0+n)*ld + k0+k].
__device__ inline v16h load_b_lds(const _Float16* m, int n0, int k0, int ld,
                                  int lane) {
  const _Float16* p = m + (n0 + (lane & 15)) * ld + k0 + ((lane >> 4) << 3);
  v16h b;
#pragma unroll
  for (int i = 0; i < 8; ++i) b[i] = p[i];
#pragma unroll
  for (int i = 0; i < 8; ++i) b[8 + i] = p[16 + i];
  return b;
}

// ---- CDNA5 data movers -----------------------------------------------------

// 16B global(f16)->LDS copy; async (ASYNCcnt) when available.
// The builtin takes typed int4 pointers (V4i*1, V4i*3): cast via integers.
__device__ inline void cp16(void* l, const void* g) {
#if defined(ASYNC_LDS)
  typedef __attribute__((address_space(1))) i32x4 gvec_t;
  typedef __attribute__((address_space(3))) i32x4 lvec_t;
  gvec_t* gp = (gvec_t*)(unsigned long long)g;
  lvec_t* lp = (lvec_t*)(unsigned long long)(unsigned)(unsigned long long)l;
  __builtin_amdgcn_global_load_async_to_lds_b128(gp, lp, 0, 0);
#else
  *(uint4*)l = *(const uint4*)g;
#endif
}

__device__ inline void wait_async_zero() {
#if defined(ASYNC_LDS)
#if __has_builtin(__builtin_amdgcn_s_wait_asynccnt)
  __builtin_amdgcn_s_wait_asynccnt(0);
#else
  asm volatile("s_wait_asynccnt 0x0" ::: "memory");
#endif
#endif
}

// TDM: load a BN x D f16 tile (row stride = D halves in global) into LDS,
// padded by 4 DWORDs every 64 DWORDs -> LDS row stride = LDK=136 halves.
// Descriptor layout per CDNA5 ISA (D# groups 0/1).
__device__ inline void tdm_load_k64(unsigned lds_addr, const void* gsrc) {
  const unsigned long long ga = (unsigned long long)gsrc;
  u32x4 g0;
  g0[0] = 1u;                                    // count=1 (valid user D#)
  g0[1] = lds_addr;                              // lds_addr (bytes)
  g0[2] = (unsigned)ga;                          // global_addr[31:0]
  g0[3] = (unsigned)((ga >> 32) & 0x01ffffffull) // global_addr[56:32]
        | (2u << 30);                            // type=2 ("image")
  i32x8 g1;
  g1[0] = (1 << 16)      // data_size = 1 -> 2-byte elements
        | (1 << 20)      // pad_enable
        | (5 << 22)      // pad_interval: 64 DWORDs (one 256B row)
        | (3 << 25);     // pad_amount: 4 DWORDs (16B) -> stride 272B
  g1[1] = (int)((unsigned)D << 16);     // tensor_dim0 = 128
  g1[2] = (int)((unsigned)S << 16);     // tensor_dim1 = 3840 (low half)
  g1[3] = (int)((unsigned)D << 16);     // tile_dim0 = 128
  g1[4] = BN;                           // tile_dim1 = 64, tile_dim2 = 0
  g1[5] = D;                            // tensor_dim0_stride = 128 (low 32)
  g1[6] = 0;                            // stride hi / dim1_stride low
  g1[7] = 0;
  const i32x4 gz = {0, 0, 0, 0};
  asm volatile("tensor_load_to_lds %0, %1, %2, %3"
               :: "s"(g0), "s"(g1), "s"(gz), "s"(gz)
               : "memory");
}

__device__ inline unsigned lds_addr_of(const void* p) {
  // generic LDS pointer = {shared aperture, offset}: low 32 bits are the LDS address
  return (unsigned)(unsigned long long)p;
}

} // namespace

// ---------------------------------------------------------------------------
// Kernel 0: one-time K fp32 -> fp16 conversion (enables TDM/async LDS fills)
// ---------------------------------------------------------------------------
__global__ __launch_bounds__(256) void cvt_k_f16_kernel(
    const float* __restrict__ x, _Float16* __restrict__ xh) {
  constexpr int N8 = H * S * D / 8;
  const int i = blockIdx.x * 256 + threadIdx.x;
  if (i >= N8) return;
  const float4 a = ((const float4*)x)[2 * i];
  const float4 b = ((const float4*)x)[2 * i + 1];
  v8h o;
  o[0] = (_Float16)a.x; o[1] = (_Float16)a.y;
  o[2] = (_Float16)a.z; o[3] = (_Float16)a.w;
  o[4] = (_Float16)b.x; o[5] = (_Float16)b.y;
  o[6] = (_Float16)b.z; o[7] = (_Float16)b.w;
  ((v8h*)xh)[i] = o;
}

// ---------------------------------------------------------------------------
// Kernel 1: dense colsum attention -> bs[g][key],  g = h*QG + qg
// K chunks arrive via TDM (double-buffered), overlapped with WMMA compute.
// ---------------------------------------------------------------------------
__global__ __launch_bounds__(NW * 32) void colsum_attn_kernel(
    const float* __restrict__ q, const _Float16* __restrict__ kh,
    float* __restrict__ bs) {
  __shared__ __align__(16) _Float16 Ks[2][BN * LDK];
  __shared__ float part[NW * BN];

  const int g    = blockIdx.x;
  const int hh   = g / QG;
  const int qg   = g % QG;
  const int tid  = threadIdx.x;
  const int wave = tid >> 5;
  const int lane = tid & 31;
  const int ln   = lane & 15;
  const int rowbase = wave * 32;
  constexpr int NC = S / BN;   // 60 chunks

  const float*    qb  = q  + ((size_t)hh * S + (size_t)qg * BM) * D;
  const _Float16* khh = kh + (size_t)hh * S * D;

  // Q fragments resident in VGPRs, pre-scaled into the exp2 domain.
  v16h qf[2][4];
#pragma unroll
  for (int rt = 0; rt < 2; ++rt)
#pragma unroll
    for (int ks = 0; ks < 4; ++ks)
      qf[rt][ks] = load_a_global(qb, rowbase + rt * 16, ks * 32, lane, SCALE_LOG2E);

  float m0[2][8], l0[2][8];
#pragma unroll
  for (int rt = 0; rt < 2; ++rt)
#pragma unroll
    for (int r = 0; r < 8; ++r) { m0[rt][r] = -1e30f; l0[rt][r] = 0.0f; }

  // ---- pass A: online row max + exp2-sum over all keys ----
  if (wave == 0) {
    tdm_load_k64(lds_addr_of(&Ks[0][0]), khh);
    __builtin_amdgcn_s_wait_tensorcnt(0);
  }
  __syncthreads();
  for (int c = 0; c < NC; ++c) {
    const int buf = c & 1;
    if (wave == 0 && c + 1 < NC)
      tdm_load_k64(lds_addr_of(&Ks[buf ^ 1][0]), khh + (size_t)(c + 1) * BN * D);
#pragma unroll
    for (int rt = 0; rt < 2; ++rt) {
      v8f st[4];
#pragma unroll
      for (int nt = 0; nt < 4; ++nt) {
        v8f acc = {};
#pragma unroll
        for (int ks = 0; ks < 4; ++ks)
          acc = wmma16(qf[rt][ks], load_b_lds(Ks[buf], nt * 16, ks * 32, LDK, lane), acc);
        st[nt] = acc;
      }
#pragma unroll
      for (int r = 0; r < 8; ++r) {
        float cm = st[0][r];
#pragma unroll
        for (int nt = 1; nt < 4; ++nt) cm = fmaxf(cm, st[nt][r]);
        cm = redmax16(cm);
        const float mn = fmaxf(m0[rt][r], cm);
        float rs = 0.0f;
#pragma unroll
        for (int nt = 0; nt < 4; ++nt) rs += exp2f(st[nt][r] - mn);
        rs = redsum16(rs);
        l0[rt][r] = l0[rt][r] * exp2f(m0[rt][r] - mn) + rs;
        m0[rt][r] = mn;
      }
    }
    if (wave == 0) __builtin_amdgcn_s_wait_tensorcnt(0);
    __syncthreads();
  }

  float li[2][8];
#pragma unroll
  for (int rt = 0; rt < 2; ++rt)
#pragma unroll
    for (int r = 0; r < 8; ++r) li[rt][r] = 1.0f / l0[rt][r];

  // ---- pass B: recompute scores, column-sum normalized probs ----
  if (wave == 0) {
    tdm_load_k64(lds_addr_of(&Ks[0][0]), khh);
    __builtin_amdgcn_s_wait_tensorcnt(0);
  }
  __syncthreads();
  for (int c = 0; c < NC; ++c) {
    const int buf = c & 1;
    if (wave == 0 && c + 1 < NC)
      tdm_load_k64(lds_addr_of(&Ks[buf ^ 1][0]), khh + (size_t)(c + 1) * BN * D);
    float cs[4] = {0.0f, 0.0f, 0.0f, 0.0f};
#pragma unroll
    for (int rt = 0; rt < 2; ++rt) {
#pragma unroll
      for (int nt = 0; nt < 4; ++nt) {
        v8f acc = {};
#pragma unroll
        for (int ks = 0; ks < 4; ++ks)
          acc = wmma16(qf[rt][ks], load_b_lds(Ks[buf], nt * 16, ks * 32, LDK, lane), acc);
        float s = 0.0f;
#pragma unroll
        for (int r = 0; r < 8; ++r)
          s += exp2f(acc[r] - m0[rt][r]) * li[rt][r];
        cs[nt] += s;   // lane's 16-row partial colsum for column nt*16+ln
      }
    }
#pragma unroll
    for (int nt = 0; nt < 4; ++nt) {
      const float t = cs[nt] + __shfl_xor(cs[nt], 16, 32);  // merge rows 0-7 / 8-15
      if (lane < 16) part[wave * BN + nt * 16 + ln] = t;
    }
    __syncthreads();
    if (tid < BN) {
      float s = 0.0f;
#pragma unroll
      for (int w = 0; w < NW; ++w) s += part[w * BN + tid];
      bs[(size_t)g * S + c * BN + tid] = s;
    }
    if (wave == 0) __builtin_amdgcn_s_wait_tensorcnt(0);
    __syncthreads();
  }
}

// ---------------------------------------------------------------------------
// Kernel 2: exact deterministic top-k via rank counting (strict total order).
// ---------------------------------------------------------------------------
__global__ __launch_bounds__(256) void topk_select_kernel(
    const float* __restrict__ bs, int* __restrict__ inds,
    const int* __restrict__ topk_p) {
  __shared__ float sb[S];
  const int g = blockIdx.x;
  const int topk = *topk_p;
  for (int i = threadIdx.x; i < S; i += 256) sb[i] = bs[(size_t)g * S + i];
  __syncthreads();
  for (int e = threadIdx.x; e < S; e += 256) {
    const float v = sb[e];
    int rank = 0;
    for (int j = 0; j < S; ++j) {
      const float vj = sb[j];
      rank += (vj > v) || (vj == v && j < e);
    }
    if (rank < topk) inds[(size_t)g * topk + rank] = e;  // rank is a permutation
  }
}

// ---------------------------------------------------------------------------
// Kernel 3: block-sparse gather flash-attention over the selected keys.
// Gathered K rows stream through global_load_async_to_lds (ASYNCcnt).
// ---------------------------------------------------------------------------
__global__ __launch_bounds__(NW * 32) void csp_attn_kernel(
    const float* __restrict__ q, const _Float16* __restrict__ kh,
    const float* __restrict__ v, const int* __restrict__ inds,
    float* __restrict__ out, const int* __restrict__ topk_p) {
  __shared__ __align__(16) _Float16 Ks[BN * LDK];   // gathered K rows, row-major f16
  __shared__ __align__(16) _Float16 Vt[D * LDV];    // gathered V, transposed [d][key]
  __shared__ __align__(16) _Float16 Ps[BM * LDP];   // staged P for PV A-fragments
  __shared__ int ii[BN];

  const int g    = blockIdx.x;
  const int hh   = g / QG;
  const int qg   = g % QG;
  const int tid  = threadIdx.x;
  const int wave = tid >> 5;
  const int lane = tid & 31;
  const int hi   = lane >> 4;
  const int ln   = lane & 15;
  const int rowbase = wave * 32;
  const int topk = *topk_p;

  const float*    qb  = q  + ((size_t)hh * S + (size_t)qg * BM) * D;
  const _Float16* khh = kh + (size_t)hh * S * D;
  const float*    vb  = v  + (size_t)hh * S * D;

  v16h qf[2][4];
#pragma unroll
  for (int rt = 0; rt < 2; ++rt)
#pragma unroll
    for (int ks = 0; ks < 4; ++ks)
      qf[rt][ks] = load_a_global(qb, rowbase + rt * 16, ks * 32, lane, SCALE_LOG2E);

  float m0[2][8], l0[2][8];
  v8f oacc[2][8];
  const v8f vzero = {};
#pragma unroll
  for (int rt = 0; rt < 2; ++rt) {
#pragma unroll
    for (int r = 0; r < 8; ++r) { m0[rt][r] = -1e30f; l0[rt][r] = 0.0f; }
#pragma unroll
    for (int ct = 0; ct < 8; ++ct) oacc[rt][ct] = vzero;
  }

  const int nch = topk / BN;
  for (int c = 0; c < nch; ++c) {
    __syncthreads();
    if (tid < BN) ii[tid] = inds[(size_t)g * topk + c * BN + tid];
    __syncthreads();
    // K rows: 16B async copies from the f16 K image straight into padded LDS.
    for (int i = tid; i < BN * (D / 8); i += NW * 32) {
      const int r  = i >> 4;           // D/8 = 16 chunks of 8 halves per row
      const int c8 = (i & 15) * 8;
      cp16(&Ks[r * LDK + c8], khh + (size_t)ii[r] * D + c8);
    }
    // V rows: fp32 load + transpose-scatter (async copy cannot transpose).
    for (int i = tid; i < BN * (D / 4); i += NW * 32) {
      const int r  = i / (D / 4);
      const int c4 = (i % (D / 4)) * 4;
      const int src = ii[r];
      const float4 vf = *(const float4*)(vb + (size_t)src * D + c4);
      Vt[(c4 + 0) * LDV + r] = (_Float16)vf.x;
      Vt[(c4 + 1) * LDV + r] = (_Float16)vf.y;
      Vt[(c4 + 2) * LDV + r] = (_Float16)vf.z;
      Vt[(c4 + 3) * LDV + r] = (_Float16)vf.w;
    }
    wait_async_zero();
    __syncthreads();

#pragma unroll
    for (int rt = 0; rt < 2; ++rt) {
      // ---- S = Q K^T (exp2 domain) ----
      v8f st[4];
#pragma unroll
      for (int nt = 0; nt < 4; ++nt) {
        v8f acc = {};
#pragma unroll
        for (int ks = 0; ks < 4; ++ks)
          acc = wmma16(qf[rt][ks], load_b_lds(Ks, nt * 16, ks * 32, LDK, lane), acc);
        st[nt] = acc;
      }
      // ---- online softmax update ----
#pragma unroll
      for (int r = 0; r < 8; ++r) {
        float cm = st[0][r];
#pragma unroll
        for (int nt = 1; nt < 4; ++nt) cm = fmaxf(cm, st[nt][r]);
        cm = redmax16(cm);
        const float mn = fmaxf(m0[rt][r], cm);
        const float al = exp2f(m0[rt][r] - mn);
#pragma unroll
        for (int ct = 0; ct < 8; ++ct) oacc[rt][ct][r] *= al;
        float rs = 0.0f;
#pragma unroll
        for (int nt = 0; nt < 4; ++nt) {
          const float e = exp2f(st[nt][r] - mn);
          st[nt][r] = e;   // st becomes unnormalized P
          rs += e;
        }
        rs = redsum16(rs);
        l0[rt][r] = l0[rt][r] * al + rs;
        m0[rt][r] = mn;
      }
      // ---- stage P as f16 in wave-private LDS rows ----
#pragma unroll
      for (int nt = 0; nt < 4; ++nt)
#pragma unroll
        for (int r = 0; r < 8; ++r)
          Ps[(rowbase + rt * 16 + hi * 8 + r) * LDP + nt * 16 + ln] =
              (_Float16)st[nt][r];
      // ---- O += P @ V  (K = 64 keys -> two 32-wide k-steps) ----
#pragma unroll
      for (int ct = 0; ct < 8; ++ct)
#pragma unroll
        for (int ks = 0; ks < 2; ++ks) {
          const v16h pa = load_a_lds(Ps, rowbase + rt * 16, ks * 32, LDP, lane);
          const v16h bv = load_b_lds(Vt, ct * 16, ks * 32, LDV, lane);
          oacc[rt][ct] = wmma16(pa, bv, oacc[rt][ct]);
        }
    }
  }

  // ---- normalize and store O (16 lanes -> 16 consecutive cols, coalesced) ----
#pragma unroll
  for (int rt = 0; rt < 2; ++rt) {
#pragma unroll
    for (int r = 0; r < 8; ++r) {
      const float inv = 1.0f / l0[rt][r];
      const int row = qg * BM + rowbase + rt * 16 + hi * 8 + r;
      float* op = out + ((size_t)hh * S + row) * D + ln;
#pragma unroll
      for (int ct = 0; ct < 8; ++ct) op[ct * 16] = oacc[rt][ct][r] * inv;
    }
  }
}

// ---------------------------------------------------------------------------
extern "C" void kernel_launch(void* const* d_in, const int* in_sizes, int n_in,
                              void* d_out, int out_size, void* d_ws, size_t ws_size,
                              hipStream_t stream) {
  (void)in_sizes; (void)n_in; (void)out_size; (void)ws_size;
  const float* q = (const float*)d_in[0];
  const float* k = (const float*)d_in[1];
  const float* v = (const float*)d_in[2];
  const int* topk = (const int*)d_in[3];   // device scalar; read inside kernels
  float* out = (float*)d_out;

  // ws layout: [0,4MiB) bs (needs 3.69MB) | [4MiB,8MiB) inds | [8MiB,...) K fp16 (11.8MB)
  float*    bs  = (float*)d_ws;
  int*      ind = (int*)((char*)d_ws + (4u << 20));
  _Float16* kH  = (_Float16*)((char*)d_ws + (8u << 20));

  const dim3 grid(H * QG);
  cvt_k_f16_kernel<<<dim3(H * S * D / 8 / 256), 256, 0, stream>>>(k, kH);
  colsum_attn_kernel<<<grid, NW * 32, 0, stream>>>(q, kH, bs);
  topk_select_kernel<<<grid, 256, 0, stream>>>(bs, ind, topk);
  csp_attn_kernel<<<grid, NW * 32, 0, stream>>>(q, kH, v, ind, out, topk);
}
